// MAB_72911364817388
// MI455X (gfx1250) — matrix operand order repaired and checked by reading, not compiled, not run on hardware
//
#include <hip/hip_runtime.h>

// ---------------------------------------------------------------------------
// CDNA5 (gfx1250) fused MAB: QKV proj (WMMA bf16) -> flash attention (WMMA)
//   -> residual + LN -> relu-MLP residual (WMMA) -> LN.  wave32 everywhere.
// Data movement: GLOBAL_LOAD_ASYNC_TO_LDS_B128 + ASYNCcnt, double-buffered.
// ---------------------------------------------------------------------------

typedef __attribute__((ext_vector_type(16))) __bf16 v16bf;
typedef __attribute__((ext_vector_type(8)))  __bf16 v8bf;
typedef __attribute__((ext_vector_type(8)))  float  v8f;

#define B_   8
#define NQ_  1024
#define NK_  1024
#define D_   512
#define H_   8
#define HD_  64
#define MROWS (B_ * NQ_)          // 8192

static __device__ __forceinline__ v8f wmma_bf16(v16bf a, v16bf b, v8f c) {
  // emits v_wmma_f32_16x16x32_bf16
  return __builtin_amdgcn_wmma_f32_16x16x32_bf16(
      /*neg_a=*/false, a, /*neg_b=*/false, b,
      /*c_mod=*/(short)0, c, /*reuse_a=*/false, /*reuse_b=*/false);
}

static __device__ __forceinline__ v16bf cat8(v8bf lo, v8bf hi) {
  v16bf r;
#pragma unroll
  for (int i = 0; i < 8; ++i) { r[i] = lo[i]; r[i + 8] = hi[i]; }
  return r;
}

// LDS byte offset of a generic shared pointer (ISA 10.2: addr[31:0] = LDS off)
static __device__ __forceinline__ uint32_t lds_off(const void* p) {
  return (uint32_t)(uintptr_t)p;
}

// async DMA: 16 bytes global -> LDS per lane, tracked by ASYNCcnt (CDNA5)
static __device__ __forceinline__ void async_b128(uint32_t ldsaddr,
                                                  const void* gaddr) {
  asm volatile("global_load_async_to_lds_b128 %0, %1, off"
               :: "v"(ldsaddr), "v"(gaddr) : "memory");
}
static __device__ __forceinline__ void wait_async0() {
  asm volatile("s_wait_asynccnt 0x0" ::: "memory");
}

static __device__ __forceinline__ float half_reduce_max(float v) {
#pragma unroll
  for (int off = 1; off < 16; off <<= 1) v = fmaxf(v, __shfl_xor(v, off, 32));
  return v;
}
static __device__ __forceinline__ float half_reduce_add(float v) {
#pragma unroll
  for (int off = 1; off < 16; off <<= 1) v += __shfl_xor(v, off, 32);
  return v;
}

// ---------------------------------------------------------------------------
// f32 -> bf16 conversion (4-wide)
// ---------------------------------------------------------------------------
__global__ void cvt_bf16_kernel(const float4* __restrict__ src,
                                __bf16* __restrict__ dst, int n4) {
  int i = blockIdx.x * blockDim.x + threadIdx.x;
  if (i < n4) {
    float4 v = src[i];
    int b = i * 4;
    dst[b + 0] = (__bf16)v.x;
    dst[b + 1] = (__bf16)v.y;
    dst[b + 2] = (__bf16)v.z;
    dst[b + 3] = (__bf16)v.w;
  }
}

// weights: W[k][n] f32 -> Wt[n][k] bf16  (so GEMM B-tiles are K-contiguous)
__global__ void cvt_wT_kernel(const float* __restrict__ W,
                              __bf16* __restrict__ Wt) {
  int i = blockIdx.x * blockDim.x + threadIdx.x;   // 512*512 threads
  int k = i & 511, n = i >> 9;
  Wt[(size_t)n * 512 + k] = (__bf16)W[(size_t)k * 512 + n];
}

// ---------------------------------------------------------------------------
// GEMM: C[M,512] = A[M,512](bf16) @ W(bf16, pre-transposed [n][k]) + bias
// epilogues: outF (f32), outB (bf16), outBT (bf16 per-head transposed for V),
//            relu: out = res + relu(C)
// Block: 256 thr = 8 waves, 128x128 tile, K step 32, double-buffered async LDS.
// ---------------------------------------------------------------------------
__global__ __launch_bounds__(256) void gemm_bias_kernel(
    const __bf16* __restrict__ A, const __bf16* __restrict__ Wt,
    const float* __restrict__ bias, const float* __restrict__ res,
    float* __restrict__ outF, __bf16* __restrict__ outB,
    __bf16* __restrict__ outBT, int relu) {
  __shared__ alignas(128) __bf16 Als[2][128][32];   // [m][k]  2 x 8 KB
  __shared__ alignas(128) __bf16 Wls[2][128][32];   // [n][k]  2 x 8 KB

  const int tid   = threadIdx.x;
  const int w     = tid >> 5;
  const int lane  = tid & 31;
  const int lmod  = lane & 15;
  const int lhalf = lane >> 4;
  const int wm    = w >> 1;        // 0..3 : 32-row group
  const int wn    = w & 1;         // 0..1 : 64-col group
  const int m0    = blockIdx.y * 128;
  const int n0    = blockIdx.x * 128;

  auto stage = [&](int buf, int kk) {
#pragma unroll
    for (int i = 0; i < 2; ++i) {
      int li = tid + i * 256;
      int row = li >> 2, seg = li & 3;           // 128 rows x 4 16B segs
      async_b128(lds_off(&Als[buf][row][seg * 8]),
                 A + (size_t)(m0 + row) * 512 + kk + seg * 8);
      async_b128(lds_off(&Wls[buf][row][seg * 8]),
                 Wt + (size_t)(n0 + row) * 512 + kk + seg * 8);
    }
  };

  v8f acc[2][4];
#pragma unroll
  for (int s = 0; s < 2; ++s)
#pragma unroll
    for (int t = 0; t < 4; ++t) acc[s][t] = (v8f){0, 0, 0, 0, 0, 0, 0, 0};

  stage(0, 0);
  wait_async0();
  __syncthreads();

  for (int kk = 0; kk < 512; kk += 32) {
    const int cur = (kk >> 5) & 1;
    if (kk + 32 < 512) stage(cur ^ 1, kk + 32);   // prefetch next tile (DMA)

    // A fragments: lane<16 row=lane, k in {0..15}; lane>=16 k+=16
    v16bf af[2];
#pragma unroll
    for (int s = 0; s < 2; ++s) {
      int row = wm * 32 + s * 16 + lmod;
      v8bf lo = *reinterpret_cast<const v8bf*>(&Als[cur][row][lhalf * 16]);
      v8bf hi = *reinterpret_cast<const v8bf*>(&Als[cur][row][lhalf * 16 + 8]);
      af[s] = cat8(lo, hi);
    }
#pragma unroll
    for (int t = 0; t < 4; ++t) {
      int col = wn * 64 + t * 16 + lmod;   // B-frag: col per lane, K contiguous
      v16bf bf = *reinterpret_cast<const v16bf*>(&Wls[cur][col][lhalf * 16]);
#pragma unroll
      for (int s = 0; s < 2; ++s) acc[s][t] = wmma_bf16(af[s], bf, acc[s][t]);
    }

    wait_async0();       // own wave's DMA into next buffer has landed
    __syncthreads();     // everyone done reading cur + sees next buffer
  }

  // epilogue
#pragma unroll
  for (int s = 0; s < 2; ++s)
#pragma unroll
    for (int t = 0; t < 4; ++t) {
      int col = n0 + wn * 64 + t * 16 + lmod;
      float bv = bias[col];
#pragma unroll
      for (int g = 0; g < 8; ++g) {
        int row = m0 + wm * 32 + s * 16 + g + 8 * lhalf;
        size_t idx = (size_t)row * 512 + col;
        float v = acc[s][t][g] + bv;
        if (relu) v = res[idx] + fmaxf(v, 0.0f);
        if (outF) outF[idx] = v;
        if (outB) outB[idx] = (__bf16)v;
        if (outBT) {     // per-head transposed: [(b*H+h)*64+d][key]
          int bb = row >> 10, key = row & 1023;
          int hh = col >> 6, dd = col & 63;
          outBT[(((size_t)(bb * H_ + hh) * HD_ + dd) << 10) + key] = (__bf16)v;
        }
      }
    }
}

// ---------------------------------------------------------------------------
// Flash attention: block = one (b,h) x 64 query rows. 128 thr = 4 waves,
// wave owns 16 rows. 32-key chunks, double-buffered async LDS, online softmax.
// out = qh(f32 residual) + softmax(qh kh^T / sqrt(512), masked) @ vh
// ---------------------------------------------------------------------------
__global__ __launch_bounds__(128) void attn_kernel(
    const __bf16* __restrict__ qb, const __bf16* __restrict__ kb,
    const __bf16* __restrict__ vT, const float* __restrict__ qf,
    const int* __restrict__ mask, float* __restrict__ outO) {
  __shared__ alignas(128) __bf16 Qs[64][64];          // 8 KB [qrow][d]
  __shared__ alignas(128) __bf16 Ks[2][32][64];       // 8 KB [key][d]
  __shared__ alignas(128) __bf16 VTs[2][64][32];      // 8 KB [d][key]
  __shared__ alignas(128) __bf16 Ps[4][16][32];       // 4 KB per-wave P
  __shared__ int Ms[2][32];

  const int tid   = threadIdx.x;
  const int w     = tid >> 5;
  const int lane  = tid & 31;
  const int lmod  = lane & 15;
  const int lhalf = lane >> 4;
  const int b     = blockIdx.x >> 3;
  const int h     = blockIdx.x & 7;
  const int q0    = blockIdx.y * 64;
  const float scale = 0.04419417382415922f;           // 1/sqrt(512)

  auto stageKV = [&](int buf, int n0) {
#pragma unroll
    for (int i = 0; i < 2; ++i) {
      int li = tid + i * 128;
      int kr = li >> 3, ks = li & 7;                  // 32 rows x 8 segs
      async_b128(lds_off(&Ks[buf][kr][ks * 8]),
                 kb + ((size_t)(b * NK_ + n0 + kr) * D_) + h * HD_ + ks * 8);
      int dr = li >> 2, vs = li & 3;                  // 64 rows x 4 segs
      async_b128(lds_off(&VTs[buf][dr][vs * 8]),
                 vT + ((size_t)((b * H_ + h) * HD_ + dr) * NK_) + n0 + vs * 8);
    }
    if (tid < 32) Ms[buf][tid] = mask[b * NK_ + n0 + tid];
  };

  // stage Q tile (64 x 64) + first K/V chunk via async DMA
#pragma unroll
  for (int i = 0; i < 4; ++i) {
    int li = tid + i * 128;
    int row = li >> 3, seg = li & 7;
    async_b128(lds_off(&Qs[row][seg * 8]),
               qb + ((size_t)(b * NQ_ + q0 + row) * D_) + h * HD_ + seg * 8);
  }
  stageKV(0, 0);
  wait_async0();
  __syncthreads();

  // per-wave A fragments of qh (d chunks 0 and 32)
  v16bf af[2];
#pragma unroll
  for (int dc = 0; dc < 2; ++dc) {
    int row = w * 16 + lmod;
    v8bf lo = *reinterpret_cast<const v8bf*>(&Qs[row][dc * 32 + lhalf * 16]);
    v8bf hi = *reinterpret_cast<const v8bf*>(&Qs[row][dc * 32 + lhalf * 16 + 8]);
    af[dc] = cat8(lo, hi);
  }

  float m[8], l[8];
  v8f o[4];
#pragma unroll
  for (int g = 0; g < 8; ++g) { m[g] = -INFINITY; l[g] = 0.0f; }
#pragma unroll
  for (int t = 0; t < 4; ++t) o[t] = (v8f){0, 0, 0, 0, 0, 0, 0, 0};

  for (int n0 = 0; n0 < NK_; n0 += 32) {
    const int cur = (n0 >> 5) & 1;
    if (n0 + 32 < NK_) stageKV(cur ^ 1, n0 + 32);     // DMA next chunk

    // S tiles: s0 = keys[0..15], s1 = keys[16..31]
    v8f s0 = (v8f){0, 0, 0, 0, 0, 0, 0, 0};
    v8f s1 = (v8f){0, 0, 0, 0, 0, 0, 0, 0};
    {
      v16bf b00 = *reinterpret_cast<const v16bf*>(&Ks[cur][lmod][lhalf * 16]);
      v16bf b01 = *reinterpret_cast<const v16bf*>(&Ks[cur][lmod][32 + lhalf * 16]);
      v16bf b10 = *reinterpret_cast<const v16bf*>(&Ks[cur][16 + lmod][lhalf * 16]);
      v16bf b11 = *reinterpret_cast<const v16bf*>(&Ks[cur][16 + lmod][32 + lhalf * 16]);
      s0 = wmma_bf16(af[0], b00, s0);
      s0 = wmma_bf16(af[1], b01, s0);
      s1 = wmma_bf16(af[0], b10, s1);
      s1 = wmma_bf16(af[1], b11, s1);
    }

    const bool mz0 = (Ms[cur][lmod] == 0);
    const bool mz1 = (Ms[cur][16 + lmod] == 0);

    // online softmax (rows striped across 16-lane halves of the C tile)
#pragma unroll
    for (int g = 0; g < 8; ++g) {
      float f0 = mz0 ? -10000.0f : s0[g] * scale;
      float f1 = mz1 ? -10000.0f : s1[g] * scale;
      float tm = half_reduce_max(fmaxf(f0, f1));
      float mn = fmaxf(m[g], tm);
      float corr = __expf(m[g] - mn);
      float p0 = __expf(f0 - mn);
      float p1 = __expf(f1 - mn);
      float rs = half_reduce_add(p0 + p1);
      l[g] = l[g] * corr + rs;
      m[g] = mn;
#pragma unroll
      for (int t = 0; t < 4; ++t) o[t][g] *= corr;
      Ps[w][8 * lhalf + g][lmod]      = (__bf16)p0;   // C-layout -> row-major
      Ps[w][8 * lhalf + g][16 + lmod] = (__bf16)p1;
    }

    // P (16x32) @ V (32x64)
    v8bf plo = *reinterpret_cast<const v8bf*>(&Ps[w][lmod][lhalf * 16]);
    v8bf phi = *reinterpret_cast<const v8bf*>(&Ps[w][lmod][lhalf * 16 + 8]);
    v16bf pf = cat8(plo, phi);
#pragma unroll
    for (int dt = 0; dt < 4; ++dt) {
      v16bf vf =
          *reinterpret_cast<const v16bf*>(&VTs[cur][dt * 16 + lmod][lhalf * 16]);
      o[dt] = wmma_bf16(pf, vf, o[dt]);
    }

    wait_async0();
    __syncthreads();
  }

  // epilogue: divide by l, add f32 qh residual, store
#pragma unroll
  for (int dt = 0; dt < 4; ++dt) {
#pragma unroll
    for (int g = 0; g < 8; ++g) {
      int row = q0 + w * 16 + g + 8 * lhalf;
      int col = h * HD_ + dt * 16 + lmod;
      size_t idx = (size_t)(b * NQ_ + row) * D_ + col;
      outO[idx] = qf[idx] + o[dt][g] / l[g];
    }
  }
}

// ---------------------------------------------------------------------------
// LayerNorm over D=512 per row; one wave per row (8 waves / block).
// ---------------------------------------------------------------------------
__global__ __launch_bounds__(256) void ln_kernel(
    const float* __restrict__ x, const float* __restrict__ gw,
    const float* __restrict__ bw, float* __restrict__ outF,
    __bf16* __restrict__ outB) {
  const int w = threadIdx.x >> 5, lane = threadIdx.x & 31;
  const int row = blockIdx.x * 8 + w;
  const float* xr = x + (size_t)row * 512;

  float vals[16];
  float s = 0.0f, sq = 0.0f;
#pragma unroll
  for (int j = 0; j < 4; ++j) {
    float4 t = *reinterpret_cast<const float4*>(xr + j * 128 + lane * 4);
    vals[j * 4 + 0] = t.x; vals[j * 4 + 1] = t.y;
    vals[j * 4 + 2] = t.z; vals[j * 4 + 3] = t.w;
    s += t.x + t.y + t.z + t.w;
    sq += t.x * t.x + t.y * t.y + t.z * t.z + t.w * t.w;
  }
#pragma unroll
  for (int off = 1; off < 32; off <<= 1) {
    s += __shfl_xor(s, off, 32);
    sq += __shfl_xor(sq, off, 32);
  }
  const float mu = s * (1.0f / 512.0f);
  const float var = sq * (1.0f / 512.0f) - mu * mu;
  const float r = rsqrtf(var + 1e-5f);
#pragma unroll
  for (int j = 0; j < 4; ++j)
#pragma unroll
    for (int c = 0; c < 4; ++c) {
      int col = j * 128 + lane * 4 + c;
      float y = (vals[j * 4 + c] - mu) * r * gw[col] + bw[col];
      size_t idx = (size_t)row * 512 + col;
      if (outF) outF[idx] = y;
      if (outB) outB[idx] = (__bf16)y;
    }
}

// ---------------------------------------------------------------------------
// host launcher
// ---------------------------------------------------------------------------
extern "C" void kernel_launch(void* const* d_in, const int* in_sizes, int n_in,
                              void* d_out, int out_size, void* d_ws,
                              size_t ws_size, hipStream_t stream) {
  (void)in_sizes; (void)n_in; (void)out_size; (void)ws_size;
  const float* Q    = (const float*)d_in[0];
  const float* K    = (const float*)d_in[1];
  const int*   msk  = (const int*)d_in[2];
  const float* Wq   = (const float*)d_in[3];
  const float* bq   = (const float*)d_in[4];
  const float* Wk   = (const float*)d_in[5];
  const float* bk   = (const float*)d_in[6];
  const float* Wv   = (const float*)d_in[7];
  const float* bv   = (const float*)d_in[8];
  const float* Wo   = (const float*)d_in[9];
  const float* bo   = (const float*)d_in[10];
  const float* g0   = (const float*)d_in[11];
  const float* b0   = (const float*)d_in[12];
  const float* g1   = (const float*)d_in[13];
  const float* b1   = (const float*)d_in[14];

  char* ws = (char*)d_ws;
  const size_t QK_E = (size_t)MROWS * D_;         // 4 Mi elems
  const size_t W_E  = (size_t)D_ * D_;            // 256 Ki elems
  size_t off = 0;
  __bf16* Qb   = (__bf16*)(ws + off); off += QK_E * 2;    //  8 MB
  __bf16* Kb   = (__bf16*)(ws + off); off += QK_E * 2;    //  8 MB
  __bf16* WqT  = (__bf16*)(ws + off); off += W_E * 2;
  __bf16* WkT  = (__bf16*)(ws + off); off += W_E * 2;
  __bf16* WvT  = (__bf16*)(ws + off); off += W_E * 2;
  __bf16* WoT  = (__bf16*)(ws + off); off += W_E * 2;     //  2 MB weights
  float*  qf   = (float*)(ws + off);  off += QK_E * 4;    // 16 MB
  __bf16* qbb  = (__bf16*)(ws + off); off += QK_E * 2;    //  8 MB
  __bf16* kbb  = (__bf16*)(ws + off); off += QK_E * 2;    //  8 MB
  __bf16* vTb  = (__bf16*)(ws + off); off += QK_E * 2;    //  8 MB (per-head T)
  float*  aO   = (float*)(ws + off);  off += QK_E * 4;    // 16 MB
  // reuse: qf -> ln1(f32); Qb -> ln1(bf16); aO -> mlp out
  float*  ln1f = qf;
  __bf16* ln1b = Qb;
  float*  mlp  = aO;

  // 1) f32 -> bf16 conversions (weights pre-transposed)
  cvt_bf16_kernel<<<(int)(QK_E / 1024), 256, 0, stream>>>((const float4*)Q, Qb,
                                                          (int)(QK_E / 4));
  cvt_bf16_kernel<<<(int)(QK_E / 1024), 256, 0, stream>>>((const float4*)K, Kb,
                                                          (int)(QK_E / 4));
  cvt_wT_kernel<<<(int)(W_E / 256), 256, 0, stream>>>(Wq, WqT);
  cvt_wT_kernel<<<(int)(W_E / 256), 256, 0, stream>>>(Wk, WkT);
  cvt_wT_kernel<<<(int)(W_E / 256), 256, 0, stream>>>(Wv, WvT);
  cvt_wT_kernel<<<(int)(W_E / 256), 256, 0, stream>>>(Wo, WoT);

  // 2) projections (WMMA): q (f32+bf16), k (bf16), v (bf16, per-head transposed)
  dim3 gg(512 / 128, MROWS / 128);  // (4, 64)
  gemm_bias_kernel<<<gg, 256, 0, stream>>>(Qb, WqT, bq, nullptr, qf, qbb,
                                           nullptr, 0);
  gemm_bias_kernel<<<gg, 256, 0, stream>>>(Kb, WkT, bk, nullptr, nullptr, kbb,
                                           nullptr, 0);
  gemm_bias_kernel<<<gg, 256, 0, stream>>>(Kb, WvT, bv, nullptr, nullptr,
                                           nullptr, vTb, 0);

  // 3) flash attention + per-head Q residual
  attn_kernel<<<dim3(B_ * H_, NQ_ / 64), 128, 0, stream>>>(qbb, kbb, vTb, qf,
                                                           msk, aO);

  // 4) LayerNorm 0 (f32 + bf16 copy for next GEMM)
  ln_kernel<<<MROWS / 8, 256, 0, stream>>>(aO, g0, b0, ln1f, ln1b);

  // 5) O = ln1 + relu(ln1 @ Wo + bo)   (WMMA, fused epilogue)
  gemm_bias_kernel<<<gg, 256, 0, stream>>>(ln1b, WoT, bo, ln1f, mlp, nullptr,
                                           nullptr, 1);

  // 6) LayerNorm 1 -> final output (f32)
  ln_kernel<<<MROWS / 8, 256, 0, stream>>>(mlp, g1, b1, (float*)d_out, nullptr);
}
